// SeltAttentionBlock_80427557585627
// MI455X (gfx1250) — compile-verified, matrix operand order checked by reading, not run on hardware
//
#include <hip/hip_runtime.h>
#include <hip/hip_bf16.h>
#include <math.h>

// ---------------------------------------------------------------------------
// Types
// ---------------------------------------------------------------------------
typedef _Float16 v16h __attribute__((ext_vector_type(16)));
typedef _Float16 v8h  __attribute__((ext_vector_type(8)));
typedef _Float16 v4h  __attribute__((ext_vector_type(4)));
typedef float    v8f  __attribute__((ext_vector_type(8)));
typedef unsigned int u32x4 __attribute__((ext_vector_type(4)));
typedef int          i32x8 __attribute__((ext_vector_type(8)));
typedef int          i32x4 __attribute__((ext_vector_type(4)));

#define B_   2
#define N_   2048
#define C_   1024
#define H_   16
#define D_   64
#define HID_ 4096
#define M_   (B_ * N_)   // 4096 activation rows

// ---------------------------------------------------------------------------
// WMMA helper: D = A(16x32 f16) x B(32x16 f16) + C(16x16 f32)
// ---------------------------------------------------------------------------
static __device__ __forceinline__ v8f wmma32(v16h a, v16h b, v8f c) {
  return __builtin_amdgcn_wmma_f32_16x16x32_f16(
      false, a, false, b, (short)0, c, false, false);
}

// Load a 16x32 f16 fragment from an LDS tile stored row-major [16][ld halves].
// CDNA5 16-bit A 16x32 layout: lanes 0-15 -> K={0..7,16..23}; lanes 16-31 ->
// K={8..15,24..31}; row = lane%16. Two 16B LDS reads per lane.
static __device__ __forceinline__ v16h load_frag(const _Float16* base, int lane, int ld) {
  const int r  = lane & 15;
  const int ko = (lane >> 4) << 3;
  const _Float16* p = base + r * ld + ko;
  v8h lo = *(const v8h*)(p);
  v8h hi = *(const v8h*)(p + 16);
  v16h f;
#pragma unroll
  for (int i = 0; i < 8; ++i) { f[i] = lo[i]; f[i + 8] = hi[i]; }
  return f;
}

// ---------------------------------------------------------------------------
// Tensor Data Mover: async load of a [tile_rows x 32] f16 tile into LDS with
// hardware padding producing a 48-half row stride (16 DW data + 8 DW pad).
// D# per CDNA5 ISA ch.8: group0 {count=1, lds_addr, global_addr, type=2},
// group1 {data_size=2B, pad_enable, pad_interval=16DW, pad_amount=8DW,
//         tensor_dim0/1, tile_dim0=32, tile_dim1, dim0_stride}.
// This toolchain exposes the 6-arg builtin (extra i32x8 group + cpol).
// ---------------------------------------------------------------------------
static __device__ __forceinline__ void tdm_load_tile_f16(
    const _Float16* lds_dst, const _Float16* gsrc,
    int tile_rows, int row_stride_elems) {
  const unsigned long long ga = (unsigned long long)(uintptr_t)gsrc;
  u32x4 g0;
  g0[0] = 1u;                                         // count=1, user mode
  g0[1] = (unsigned)(uintptr_t)lds_dst;               // LDS byte address (addr[31:0])
  g0[2] = (unsigned)ga;                               // global_addr[31:0]
  g0[3] = (unsigned)((ga >> 32) & 0x01ffffffull)      // global_addr[56:32]
          | (2u << 30);                               // type=2 ("image")
  const unsigned dim0 = (unsigned)row_stride_elems;   // tensor width (elems)
  const unsigned dim1 = 1u << 20;                     // tall enough: no OOB
  i32x8 g1;
  g1[0] = (int)((1u << 16)        // data_size = 2 bytes
              | (1u << 20)        // pad_enable
              | (3u << 22)        // pad_interval: 16 DWORDs
              | (7u << 25));      // pad_amount:    8 DWORDs
  g1[1] = (int)((dim0 & 0xffffu) << 16);                       // tensor_dim0 lo16
  g1[2] = (int)((dim0 >> 16) | ((dim1 & 0xffffu) << 16));      // dim0 hi / dim1 lo
  g1[3] = (int)((dim1 >> 16) | (32u << 16));                   // dim1 hi / tile_dim0=32
  g1[4] = (int)(tile_rows & 0xffff);                           // tile_dim1 (tile_dim2=0)
  g1[5] = (int)(unsigned)row_stride_elems;                     // tensor_dim0_stride lo32
  g1[6] = 0;                                                   // stride hi / dim1_stride lo
  g1[7] = 0;
  const i32x4 gz4 = {0, 0, 0, 0};
  const i32x8 gz8 = {0, 0, 0, 0, 0, 0, 0, 0};
  __builtin_amdgcn_tensor_load_to_lds(g0, g1, gz4, gz4, gz8, 0);
}

// ---------------------------------------------------------------------------
// Weight transpose + convert:  in f32 [R][Cc] -> out f16 [Cc][R]
// ---------------------------------------------------------------------------
__global__ __launch_bounds__(256) void k_transpose_f16(const float* __restrict__ in,
                                                       _Float16* __restrict__ out,
                                                       int R, int Cc) {
  __shared__ float tile[32][33];
  const int c0 = blockIdx.x * 32, r0 = blockIdx.y * 32;
  const int tx = threadIdx.x, ty = threadIdx.y;   // block (32,8)
#pragma unroll
  for (int i = 0; i < 4; ++i) {
    int r = r0 + ty + i * 8;
    tile[ty + i * 8][tx] = in[(size_t)r * Cc + c0 + tx];
  }
  __syncthreads();
#pragma unroll
  for (int i = 0; i < 4; ++i) {
    int c = c0 + ty + i * 8;
    out[(size_t)c * R + r0 + tx] = (_Float16)tile[tx][ty + i * 8];
  }
}

// ---------------------------------------------------------------------------
// LayerNorm (fp32 math) -> f16 output. One 256-thread block per 1024-row.
// ---------------------------------------------------------------------------
__global__ __launch_bounds__(256) void k_layernorm_f16(const float* __restrict__ x,
                                                       const float* __restrict__ g,
                                                       const float* __restrict__ bta,
                                                       _Float16* __restrict__ y) {
  const int row = blockIdx.x;
  const int t = threadIdx.x;
  const float* xr = x + (size_t)row * C_;
  float4 v = *(const float4*)(xr + t * 4);
  float s  = v.x + v.y + v.z + v.w;
  float s2 = v.x * v.x + v.y * v.y + v.z * v.z + v.w * v.w;
  __shared__ float r1[256], r2[256];
  r1[t] = s; r2[t] = s2;
  __syncthreads();
  for (int st = 128; st > 0; st >>= 1) {
    if (t < st) { r1[t] += r1[t + st]; r2[t] += r2[t + st]; }
    __syncthreads();
  }
  const float mu  = r1[0] * (1.0f / C_);
  const float var = r2[0] * (1.0f / C_) - mu * mu;
  const float rs  = rsqrtf(var + 1e-5f);
  const int c = t * 4;
  v4h o;
  o[0] = (_Float16)((v.x - mu) * rs * g[c + 0] + bta[c + 0]);
  o[1] = (_Float16)((v.y - mu) * rs * g[c + 1] + bta[c + 1]);
  o[2] = (_Float16)((v.z - mu) * rs * g[c + 2] + bta[c + 2]);
  o[3] = (_Float16)((v.w - mu) * rs * g[c + 3] + bta[c + 3]);
  *(v4h*)(y + (size_t)row * C_ + c) = o;
}

// ---------------------------------------------------------------------------
// Tiled WMMA GEMM with TDM double-buffered staging.
// out[M][N] = A[M][K](f16) * Bt[N][K](f16)^T  (+ epilogue)
// Block = 128 threads (4 waves). Tile 128x64; each wave owns a 32x64 strip
// (2 A-frags x 4 B-frags -> 8 WMMAs per 32-wide K step).
// MODE 0: f16 out, no bias          (QKV)
// MODE 1: f32 out, +bias +residual  (proj / fc2)
// MODE 2: f16 out, gelu(acc+bias)   (fc1)
// ---------------------------------------------------------------------------
#define LDK 48   // 96B LDS stride: produced by TDM pad, 16B aligned

template <int MODE>
__global__ __launch_bounds__(128) void k_gemm128(const _Float16* __restrict__ A,
                                                 const _Float16* __restrict__ Bt,
                                                 const float* __restrict__ bias,
                                                 const float* __restrict__ resid,
                                                 float* __restrict__ outf,
                                                 _Float16* __restrict__ outh,
                                                 int M, int N, int K) {
  __shared__ __align__(16) _Float16 As[2][128 * LDK];
  __shared__ __align__(16) _Float16 Bs[2][64 * LDK];
  const int mBase = blockIdx.y * 128;
  const int nBase = blockIdx.x * 64;
  const int t = threadIdx.x, lane = t & 31, w = t >> 5;

  v8f acc[2][4];
  const v8f zero = {0.f, 0.f, 0.f, 0.f, 0.f, 0.f, 0.f, 0.f};
#pragma unroll
  for (int ms = 0; ms < 2; ++ms)
#pragma unroll
    for (int nt = 0; nt < 4; ++nt) acc[ms][nt] = zero;

  // Prologue: wave 0 DMAs the first K-tile pair, waits TENSORcnt, barrier.
  if (t < 32) {
    tdm_load_tile_f16(As[0], A  + (size_t)mBase * K, 128, K);
    tdm_load_tile_f16(Bs[0], Bt + (size_t)nBase * K,  64, K);
    __builtin_amdgcn_s_wait_tensorcnt(0);
  }
  __syncthreads();

  int ping = 0;
  for (int kc = 0; kc < K; kc += 32) {
    const int nxt = ping ^ 1;
    const bool more = (kc + 32) < K;
    if (more && t < 32) {      // async: DMA next tiles while we compute
      tdm_load_tile_f16(As[nxt], A  + (size_t)mBase * K + kc + 32, 128, K);
      tdm_load_tile_f16(Bs[nxt], Bt + (size_t)nBase * K + kc + 32,  64, K);
    }
    const v16h a0 = load_frag(As[ping] + (w * 32 +  0) * LDK, lane, LDK);
    const v16h a1 = load_frag(As[ping] + (w * 32 + 16) * LDK, lane, LDK);
#pragma unroll
    for (int nt = 0; nt < 4; ++nt) {
      const v16h bf = load_frag(Bs[ping] + (nt * 16) * LDK, lane, LDK);
      acc[0][nt] = wmma32(a0, bf, acc[0][nt]);
      acc[1][nt] = wmma32(a1, bf, acc[1][nt]);
    }
    if (more && t < 32) __builtin_amdgcn_s_wait_tensorcnt(0);
    __syncthreads();
    ping = nxt;
  }

  // Epilogue. C/D layout: VGPR i, lane l -> row = i + 8*(l>=16), col = l%16.
  const int rofs = (lane >> 4) << 3;
  const int cofs = lane & 15;
#pragma unroll
  for (int ms = 0; ms < 2; ++ms) {
#pragma unroll
    for (int nt = 0; nt < 4; ++nt) {
#pragma unroll
      for (int i = 0; i < 8; ++i) {
        const int row = mBase + w * 32 + ms * 16 + i + rofs;
        const int col = nBase + nt * 16 + cofs;
        float v = acc[ms][nt][i];
        if (MODE == 0) {
          outh[(size_t)row * N + col] = (_Float16)v;
        } else if (MODE == 1) {
          v += bias[col] + resid[(size_t)row * N + col];
          outf[(size_t)row * N + col] = v;
        } else {
          v += bias[col];
          v = 0.5f * v * (1.0f + erff(v * 0.70710678118654752f));  // exact GELU
          outh[(size_t)row * N + col] = (_Float16)v;
        }
      }
    }
  }
}

// ---------------------------------------------------------------------------
// Flash attention: one block per (b, h, 64-query-row tile). 4 waves.
// qkv layout: [B*N][3*C], inner [3][H][D]. Output o: f16 [B*N][C].
// ---------------------------------------------------------------------------
#define LDQ 72   // 144B stride: 16B aligned, bank-friendly

__global__ __launch_bounds__(128) void k_attention(const _Float16* __restrict__ qkv,
                                                   _Float16* __restrict__ o) {
  __shared__ __align__(16) _Float16 Qs[64 * LDQ];
  __shared__ __align__(16) _Float16 Ks[64 * LDQ];
  __shared__ __align__(16) _Float16 Vs[64 * LDQ];
  __shared__ __align__(16) _Float16 Ps[64 * LDQ];

  const int bh = blockIdx.x;         // 0..31
  const int qb = blockIdx.y;         // 0..31
  const int b  = bh >> 4;
  const int h  = bh & 15;
  const int t    = threadIdx.x;
  const int lane = t & 31;
  const int w    = t >> 5;

  const size_t rowStride = 3 * C_;
  const _Float16* qbase = qkv + (size_t)b * N_ * rowStride + 0 * C_ + h * D_;
  const _Float16* kbase = qkv + (size_t)b * N_ * rowStride + 1 * C_ + h * D_;
  const _Float16* vbase = qkv + (size_t)b * N_ * rowStride + 2 * C_ + h * D_;

  const int srow = t >> 1;
  const int scb  = (t & 1) * 32;     // 32 halves (64B) per thread

  // Stage Q tile once: [64 rows][64 d]
  {
    const _Float16* src = qbase + (size_t)(qb * 64 + srow) * rowStride + scb;
    _Float16* dst = Qs + srow * LDQ + scb;
#pragma unroll
    for (int i = 0; i < 4; ++i)
      *(uint4*)(dst + 8 * i) = *(const uint4*)(src + 8 * i);
  }

  const v8f zero = {0.f, 0.f, 0.f, 0.f, 0.f, 0.f, 0.f, 0.f};
  v8f oacc[4];
#pragma unroll
  for (int nt = 0; nt < 4; ++nt) oacc[nt] = zero;
  float mrow[8], lrow[8];
#pragma unroll
  for (int i = 0; i < 8; ++i) { mrow[i] = -1e30f; lrow[i] = 0.f; }

  const float scale = 0.125f;        // 1/sqrt(64)
  const int rofs = (lane >> 4) << 3;
  const int cofs = lane & 15;

  for (int j = 0; j < N_ / 64; ++j) {
    __syncthreads();
    {   // Stage K block: rows = key, cols = d
      const _Float16* src = kbase + (size_t)(j * 64 + srow) * rowStride + scb;
      _Float16* dst = Ks + srow * LDQ + scb;
#pragma unroll
      for (int i = 0; i < 4; ++i)
        *(uint4*)(dst + 8 * i) = *(const uint4*)(src + 8 * i);
    }
    {   // Stage V block transposed: Vs[d][key]
      const _Float16* src = vbase + (size_t)(j * 64 + srow) * rowStride + scb;
      v8h t4[4];
#pragma unroll
      for (int i = 0; i < 4; ++i) t4[i] = *(const v8h*)(src + 8 * i);
#pragma unroll
      for (int i = 0; i < 32; ++i)
        Vs[(scb + i) * LDQ + srow] = t4[i >> 3][i & 7];
    }
    if (j + 1 < N_ / 64) {           // gfx1250 global_prefetch_b8 of next block
      __builtin_prefetch(kbase + (size_t)(j * 64 + 64 + srow) * rowStride + scb, 0, 3);
      __builtin_prefetch(vbase + (size_t)(j * 64 + 64 + srow) * rowStride + scb, 0, 3);
    }
    __syncthreads();

    // S = Q * K^T  (16x64 per wave; K-dim = D = 64 -> 2 WMMA steps)
    const v16h a0 = load_frag(Qs + (w * 16) * LDQ + 0,  lane, LDQ);
    const v16h a1 = load_frag(Qs + (w * 16) * LDQ + 32, lane, LDQ);
    v8f sacc[4];
#pragma unroll
    for (int nt = 0; nt < 4; ++nt) {
      const v16h b0 = load_frag(Ks + (nt * 16) * LDQ + 0,  lane, LDQ);
      const v16h b1 = load_frag(Ks + (nt * 16) * LDQ + 32, lane, LDQ);
      v8f s = zero;
      s = wmma32(a0, b0, s);
      s = wmma32(a1, b1, s);
      sacc[nt] = s;
    }

    // Online softmax. VGPR index i owns one row per 16-lane half.
#pragma unroll
    for (int i = 0; i < 8; ++i) {
      float v = sacc[0][i] * scale;
      sacc[0][i] = v;
#pragma unroll
      for (int nt = 1; nt < 4; ++nt) {
        float u = sacc[nt][i] * scale;
        sacc[nt][i] = u;
        v = fmaxf(v, u);
      }
#pragma unroll
      for (int off = 1; off < 16; off <<= 1)     // stay inside 16-lane half
        v = fmaxf(v, __shfl_xor(v, off, 32));
      const float mnew  = fmaxf(mrow[i], v);
      const float alpha = __expf(mrow[i] - mnew);
      float rs = 0.f;
#pragma unroll
      for (int nt = 0; nt < 4; ++nt) {
        const float p = __expf(sacc[nt][i] - mnew);
        sacc[nt][i] = p;
        rs += p;
      }
#pragma unroll
      for (int off = 1; off < 16; off <<= 1)
        rs += __shfl_xor(rs, off, 32);
      lrow[i] = lrow[i] * alpha + rs;
      mrow[i] = mnew;
#pragma unroll
      for (int nt = 0; nt < 4; ++nt) oacc[nt][i] *= alpha;
    }

    // P (C/D layout) -> LDS (A layout); same-wave LDS ordering via DScnt.
#pragma unroll
    for (int nt = 0; nt < 4; ++nt)
#pragma unroll
      for (int i = 0; i < 8; ++i)
        Ps[(w * 16 + i + rofs) * LDQ + nt * 16 + cofs] = (_Float16)sacc[nt][i];

    // O += P * V  (K-dim = 64 keys -> 2 WMMA steps)
    const v16h pa0 = load_frag(Ps + (w * 16) * LDQ + 0,  lane, LDQ);
    const v16h pa1 = load_frag(Ps + (w * 16) * LDQ + 32, lane, LDQ);
#pragma unroll
    for (int nt = 0; nt < 4; ++nt) {
      const v16h bv0 = load_frag(Vs + (nt * 16) * LDQ + 0,  lane, LDQ);
      const v16h bv1 = load_frag(Vs + (nt * 16) * LDQ + 32, lane, LDQ);
      oacc[nt] = wmma32(pa0, bv0, oacc[nt]);
      oacc[nt] = wmma32(pa1, bv1, oacc[nt]);
    }
  }

  // Normalize and write o[b*N + row][h*D + d] (f16, feeds the proj GEMM).
#pragma unroll
  for (int i = 0; i < 8; ++i) {
    const float inv = 1.0f / lrow[i];
    const int row = qb * 64 + w * 16 + i + rofs;
#pragma unroll
    for (int nt = 0; nt < 4; ++nt) {
      const int col = nt * 16 + cofs;
      o[((size_t)b * N_ + row) * C_ + h * D_ + col] = (_Float16)(oacc[nt][i] * inv);
    }
  }
}

// ---------------------------------------------------------------------------
// Host-side orchestration
// ---------------------------------------------------------------------------
extern "C" void kernel_launch(void* const* d_in, const int* in_sizes, int n_in,
                              void* d_out, int out_size, void* d_ws, size_t ws_size,
                              hipStream_t stream) {
  (void)in_sizes; (void)n_in; (void)out_size; (void)ws_size;
  const float* x      = (const float*)d_in[0];
  const float* ln1_g  = (const float*)d_in[1];
  const float* ln1_b  = (const float*)d_in[2];
  const float* w_qkv  = (const float*)d_in[3];
  const float* w_proj = (const float*)d_in[4];
  const float* b_proj = (const float*)d_in[5];
  const float* ln2_g  = (const float*)d_in[6];
  const float* ln2_b  = (const float*)d_in[7];
  const float* w_fc1  = (const float*)d_in[8];
  const float* b_fc1  = (const float*)d_in[9];
  const float* w_fc2  = (const float*)d_in[10];
  const float* b_fc2  = (const float*)d_in[11];
  float* out = (float*)d_out;

  size_t off = 0;
  auto carve = [&](size_t bytes) {
    void* p = (char*)d_ws + off;
    off += (bytes + 255) & ~(size_t)255;
    return p;
  };
  _Float16* wqkv_t = (_Float16*)carve((size_t)(3 * C_) * C_ * 2);
  _Float16* wproj_t= (_Float16*)carve((size_t)C_ * C_ * 2);
  _Float16* wfc1_t = (_Float16*)carve((size_t)HID_ * C_ * 2);
  _Float16* wfc2_t = (_Float16*)carve((size_t)C_ * HID_ * 2);
  _Float16* xn1    = (_Float16*)carve((size_t)M_ * C_ * 2);
  _Float16* qkvb   = (_Float16*)carve((size_t)M_ * 3 * C_ * 2);
  _Float16* obuf   = (_Float16*)carve((size_t)M_ * C_ * 2);
  float*    x1     = (float*)   carve((size_t)M_ * C_ * 4);
  _Float16* xn2    = (_Float16*)carve((size_t)M_ * C_ * 2);
  _Float16* hbuf   = (_Float16*)carve((size_t)M_ * HID_ * 2);

  const dim3 tb(32, 8);
  k_transpose_f16<<<dim3(3 * C_ / 32, C_ / 32), tb, 0, stream>>>(w_qkv,  wqkv_t,  C_,  3 * C_);
  k_transpose_f16<<<dim3(C_ / 32,     C_ / 32), tb, 0, stream>>>(w_proj, wproj_t, C_,  C_);
  k_transpose_f16<<<dim3(HID_ / 32,   C_ / 32), tb, 0, stream>>>(w_fc1,  wfc1_t,  C_,  HID_);
  k_transpose_f16<<<dim3(C_ / 32,   HID_ / 32), tb, 0, stream>>>(w_fc2,  wfc2_t,  HID_, C_);

  k_layernorm_f16<<<M_, 256, 0, stream>>>(x, ln1_g, ln1_b, xn1);

  // qkv = xn1 @ w_qkv (f16 out)
  k_gemm128<0><<<dim3(3 * C_ / 64, M_ / 128), 128, 0, stream>>>(
      xn1, wqkv_t, nullptr, nullptr, nullptr, qkvb, M_, 3 * C_, C_);

  k_attention<<<dim3(B_ * H_, N_ / 64), 128, 0, stream>>>(qkvb, obuf);

  // x1 = x + obuf @ w_proj + b_proj (f32 out)
  k_gemm128<1><<<dim3(C_ / 64, M_ / 128), 128, 0, stream>>>(
      obuf, wproj_t, b_proj, x, x1, nullptr, M_, C_, C_);

  k_layernorm_f16<<<M_, 256, 0, stream>>>(x1, ln2_g, ln2_b, xn2);

  // h = gelu(xn2 @ w_fc1 + b_fc1) (f16 out)
  k_gemm128<2><<<dim3(HID_ / 64, M_ / 128), 128, 0, stream>>>(
      xn2, wfc1_t, b_fc1, nullptr, nullptr, hbuf, M_, HID_, C_);

  // out = x1 + h @ w_fc2 + b_fc2 (f32 out)
  k_gemm128<1><<<dim3(C_ / 64, M_ / 128), 128, 0, stream>>>(
      hbuf, wfc2_t, b_fc2, x1, out, nullptr, M_, C_, HID_);
}
